// TokenClassificationCrfHead_47510928229125
// MI455X (gfx1250) — matrix operand order repaired and checked by reading, not compiled, not run on hardware
//
#include <hip/hip_runtime.h>

typedef __attribute__((ext_vector_type(16))) _Float16 v16h;
typedef __attribute__((ext_vector_type(8)))  _Float16 v8h;
typedef __attribute__((ext_vector_type(8)))  float    v8f;

#define B_ 256
#define T_ 1024
#define K_ 128

// One workgroup = 16 batch rows, 8 waves (256 threads).
// Wave w owns output columns [16w, 16w+16).
// Per time step: row-max -> exp (f16, LDS) -> 4x v_wmma_f32_16x16x32_f16 -> log + shifts + emission.
__global__ __launch_bounds__(256) void crf_forward_kernel(
    const float* __restrict__ hs,     // [B, T, K] emissions
    const float* __restrict__ trans,  // [K, K]
    const float* __restrict__ start,  // [K]
    const float* __restrict__ endt,   // [K]
    float* __restrict__ out)          // [B]
{
  __shared__ __attribute__((aligned(16))) float    alphaS[16][K_];   // 8 KB
  __shared__ __attribute__((aligned(16))) _Float16 expAS[16][K_];    // 4 KB
  __shared__ __attribute__((aligned(16))) _Float16 Pcol[K_][K_];     // 32 KB, [n][k] = exp(trans[k][n]-c[n])
  __shared__ float cMaxS[K_];
  __shared__ float mS[16];

  const int tid  = threadIdx.x;
  const int lane = tid & 31;
  const int wave = tid >> 5;        // 0..7 -> N tile
  const int b0   = blockIdx.x * 16;

  // ---- setup: column max of transitions + P = exp(trans - cmax), column-major f16 ----
  if (tid < K_) {
    const int j = tid;
    float cm = -3.4e38f;
    for (int i = 0; i < K_; ++i) cm = fmaxf(cm, trans[i * K_ + j]);
    cMaxS[j] = cm;
    for (int i = 0; i < K_; ++i)
      Pcol[j][i] = (_Float16)__expf(trans[i * K_ + j] - cm);
  }

  // ---- alpha0 = start + emis[:, 0, :] ----
  {
    const int r  = tid >> 4;
    const int jb = (tid & 15) * 8;
    const long base = (long)(b0 + r) * T_ * K_;
    for (int e = 0; e < 8; ++e)
      alphaS[r][jb + e] = start[jb + e] + hs[base + jb + e];
  }
  __syncthreads();

  // ---- per-wave B fragments (time-invariant): 32x16 f16 tiles ----
  // B layout: lane holds column n = 16*wave + (lane&15); halves are K-contiguous,
  // K base = 16*(lane>>4) within each 32-deep chunk.
  const int ncol  = 16 * wave + (lane & 15);
  const int kbase = 16 * (lane >> 4);
  v16h bfrag[4];
  for (int kc = 0; kc < 4; ++kc)
    bfrag[kc] = *(const v16h*)&Pcol[ncol][32 * kc + kbase];

  // A layout: lane holds row M = lane&15; per 32-deep chunk the lane's 16 halves are
  // K = 16*(lane>>4) + [0..15] (two 16B chunks).
  const int arow  = lane & 15;
  const int hsel8 = 8 * (lane >> 4);

  for (int t = 1; t < T_; ++t) {
    // (a) per-row max: wave w reduces rows 2w and 2w+1 with wave32 shuffles
    for (int rr = 0; rr < 2; ++rr) {
      const int r = 2 * wave + rr;
      float v = fmaxf(fmaxf(alphaS[r][lane],      alphaS[r][lane + 32]),
                      fmaxf(alphaS[r][lane + 64], alphaS[r][lane + 96]));
      for (int off = 16; off >= 1; off >>= 1)
        v = fmaxf(v, __shfl_xor(v, off));
      if (lane == 0) mS[r] = v;
    }
    __syncthreads();

    // (b) expA = exp(alpha - m) as f16 into LDS (values in (0,1])
    {
      const int r  = tid >> 4;
      const int jb = (tid & 15) * 8;
      const float mr = mS[r];
      for (int e = 0; e < 8; ++e)
        expAS[r][jb + e] = (_Float16)__expf(alphaS[r][jb + e] - mr);
    }
    __syncthreads();

    // (c) 16x16 = (16x128)x(128x16) via 4 chained f16 WMMAs, f32 accumulate
    v8f acc = {};
    for (int kc = 0; kc < 4; ++kc) {
      v8h lo = *(const v8h*)&expAS[arow][32 * kc + 2 * hsel8];
      v8h hi = *(const v8h*)&expAS[arow][32 * kc + 2 * hsel8 + 8];
      v16h a;
      for (int e = 0; e < 8; ++e) { a[e] = lo[e]; a[8 + e] = hi[e]; }
      acc = __builtin_amdgcn_wmma_f32_16x16x32_f16(
          /*neg_a=*/false, a, /*neg_b=*/false, bfrag[kc],
          /*c_mod=*/(short)0, acc, /*reuse_a=*/false, /*reuse_b=*/false);
    }

    // (d) alpha_new[M][n] = log(acc) + m[M] + c[n] + emis[b0+M][t][n]
    // C/D layout: lane holds n = 16*wave + (lane&15); VGPR v -> M = v + 8*(lane>>4).
    {
      const float cn   = cMaxS[ncol];
      const int   mofs = 8 * (lane >> 4);
      const long  tofs = (long)t * K_ + ncol;
      float na[8];
      for (int v = 0; v < 8; ++v) {
        const int M = v + mofs;
        na[v] = __logf(acc[v]) + mS[M] + cn
              + hs[(long)(b0 + M) * T_ * K_ + tofs];
      }
      // Safe to overwrite alphaS: last reads of old alpha were before the
      // post-exp barrier; mS is only rewritten after the loop-end barrier.
      for (int v = 0; v < 8; ++v) alphaS[v + mofs][ncol] = na[v];
    }
    __syncthreads();
  }

  // ---- logZ = logsumexp_j(alpha_T + end) per batch row ----
  for (int rr = 0; rr < 2; ++rr) {
    const int r = 2 * wave + rr;
    float a0 = alphaS[r][lane]      + endt[lane];
    float a1 = alphaS[r][lane + 32] + endt[lane + 32];
    float a2 = alphaS[r][lane + 64] + endt[lane + 64];
    float a3 = alphaS[r][lane + 96] + endt[lane + 96];
    float mm = fmaxf(fmaxf(a0, a1), fmaxf(a2, a3));
    for (int off = 16; off >= 1; off >>= 1)
      mm = fmaxf(mm, __shfl_xor(mm, off));
    float s = __expf(a0 - mm) + __expf(a1 - mm) + __expf(a2 - mm) + __expf(a3 - mm);
    for (int off = 16; off >= 1; off >>= 1)
      s += __shfl_xor(s, off);
    if (lane == 0) out[b0 + r] = mm + __logf(s);
  }
}

extern "C" void kernel_launch(void* const* d_in, const int* in_sizes, int n_in,
                              void* d_out, int out_size, void* d_ws, size_t ws_size,
                              hipStream_t stream) {
  const float* hs    = (const float*)d_in[0];  // hidden_states [B,T,K] f32
  const float* trans = (const float*)d_in[1];  // transitions [K,K] f32
  const float* start = (const float*)d_in[2];  // start_transitions [K] f32
  const float* endt  = (const float*)d_in[3];  // end_transitions [K] f32
  float* out = (float*)d_out;                  // [B] f32

  crf_forward_kernel<<<B_ / 16, 256, 0, stream>>>(hs, trans, start, endt, out);
}
